// Self_Attention_24816321036860
// MI455X (gfx1250) — matrix operand order repaired and checked by reading, not compiled, not run on hardware
//
#include <hip/hip_runtime.h>
#include <hip/hip_bf16.h>

// ---------------- CDNA5 WMMA types ----------------
typedef __attribute__((ext_vector_type(16))) __bf16 v16bf;
typedef __attribute__((ext_vector_type(8)))  __bf16 v8bf;
typedef __attribute__((ext_vector_type(8)))  float  v8f;
typedef __attribute__((ext_vector_type(4)))  unsigned int u32x4;
typedef __attribute__((ext_vector_type(8)))  int i32x8;
typedef __attribute__((ext_vector_type(4)))  int i32x4;

#define B_DIM 8
#define S_DIM 2048
#define H_DIM 1024
#define M_DIM (B_DIM * S_DIM)   // 16384

#if __has_builtin(__builtin_amdgcn_tensor_load_to_lds) && \
    __has_builtin(__builtin_amdgcn_s_wait_tensorcnt)
#define USE_TDM 1
#else
#define USE_TDM 0
#endif

// WMMA bf16 fragment layout (wave32), per CDNA5 ISA §7.12.2:
//  A (16x32, MxK): row = lane&15; kb8 = (lane<16)?0:8;
//                  elems 0..7  -> K = kb8+i ; elems 8..15 -> K = kb8+16+i
//  B (32x16, KxN): col = lane&15; same K order as A; code uses kb16=(lane<16)?0:16, K = kb16+i for A and B
//  C/D (16x16 f32): elem i -> row = i + ((lane<16)?0:8); col = lane&15

static __device__ inline v8f wmma_acc(v16bf a, v16bf b, v8f c) {
    return __builtin_amdgcn_wmma_f32_16x16x32_bf16(
        false, a, false, b, (short)0, c, false, false);
}

// A fragment from bf16 memory/LDS: p at K = kb16; 16 contiguous elements.
static __device__ inline v16bf load_a_frag_bf16(const __bf16* p) {
    v8bf lo = *(const v8bf*)p;
    v8bf hi = *(const v8bf*)(p + 8);
    v16bf a;
#pragma unroll
    for (int i = 0; i < 8; ++i) { a[i] = lo[i]; a[8 + i] = hi[i]; }
    return a;
}

// B fragment from bf16 memory/LDS: 16 contiguous elements at K = kb16.
static __device__ inline v16bf load_b_frag_bf16(const __bf16* p) {
    v8bf lo = *(const v8bf*)p;
    v8bf hi = *(const v8bf*)(p + 8);
    v16bf b;
#pragma unroll
    for (int i = 0; i < 8; ++i) { b[i] = lo[i]; b[8 + i] = hi[i]; }
    return b;
}

// ============================================================================
// Kernel 0: one-shot fp32 -> bf16 conversion (8 elements / thread).
// ============================================================================
__global__ void cvt_bf16_kernel(const float* __restrict__ in,
                                __bf16* __restrict__ out, int n8) {
    const int idx = blockIdx.x * blockDim.x + threadIdx.x;
    if (idx >= n8) return;
    const float* p = in + (size_t)idx * 8;
    __bf16* o = out + (size_t)idx * 8;
#pragma unroll
    for (int i = 0; i < 8; ++i) o[i] = (__bf16)p[i];
}

// ============================================================================
// Kernel 1: y = x @ W^T + bias (all bf16 operands, f32 accum, bf16 out).
//   transposed==0: out[m*H + n]          (q, k)
//   transposed==1: out[(b*H + n)*S + s]  (v stored as vT[B][H][S])
// Block = 128 threads (4 waves); block tile 64(M) x 64(N); wave = 16 rows.
// ============================================================================
__global__ void qkv_proj_kernel(const __bf16* __restrict__ x,
                                const __bf16* __restrict__ W,
                                const float* __restrict__ bias,
                                __bf16* __restrict__ out,
                                int transposed) {
    const int tid  = threadIdx.x;
    const int wave = tid >> 5;
    const int lane = tid & 31;
    const int l15  = lane & 15;
    const int half = (lane < 16) ? 0 : 1;

    const int m0 = blockIdx.x * 64 + wave * 16;
    const int n0 = blockIdx.y * 64;

    v8f acc[4];
#pragma unroll
    for (int t = 0; t < 4; ++t)
#pragma unroll
        for (int i = 0; i < 8; ++i) acc[t][i] = 0.0f;

    const __bf16* xrow = x + (size_t)(m0 + l15) * H_DIM + half * 16;
    const __bf16* w0 = W + (size_t)(n0 +  0 + l15) * H_DIM + half * 16;
    const __bf16* w1 = W + (size_t)(n0 + 16 + l15) * H_DIM + half * 16;
    const __bf16* w2 = W + (size_t)(n0 + 32 + l15) * H_DIM + half * 16;
    const __bf16* w3 = W + (size_t)(n0 + 48 + l15) * H_DIM + half * 16;

    for (int kk = 0; kk < H_DIM; kk += 32) {
        // independent temporaries -> loads can batch under one wait
        v16bf a  = load_a_frag_bf16(xrow + kk);
        v16bf b0 = load_b_frag_bf16(w0 + kk);
        v16bf b1 = load_b_frag_bf16(w1 + kk);
        v16bf b2 = load_b_frag_bf16(w2 + kk);
        v16bf b3 = load_b_frag_bf16(w3 + kk);
        acc[0] = wmma_acc(a, b0, acc[0]);
        acc[1] = wmma_acc(a, b1, acc[1]);
        acc[2] = wmma_acc(a, b2, acc[2]);
        acc[3] = wmma_acc(a, b3, acc[3]);
    }

#pragma unroll
    for (int t = 0; t < 4; ++t) {
        const int n = n0 + t * 16 + l15;
        const float bv = bias[n];
#pragma unroll
        for (int i = 0; i < 8; ++i) {
            const int row = m0 + i + half * 8;
            const float v = acc[t][i] + bv;
            if (!transposed) {
                out[(size_t)row * H_DIM + n] = (__bf16)v;
            } else {
                const int bb = row >> 11;
                const int ss = row & (S_DIM - 1);
                out[((size_t)bb * H_DIM + n) * S_DIM + ss] = (__bf16)v;
            }
        }
    }
}

// ============================================================================
// TDM helper: 2-D tile load Global -> LDS (data_size = 2 bytes).
// D# bitfields per CDNA5 ISA.
// This toolchain exposes the 6-arg builtin:
//   (uint32x4 g0, int32x8 g1, int32x4 g2, int32x4 g3, int32x8 extra, i32 cpol)
// Groups 2/3 and the extra vector are zero (2-D tensor, no iterate/gather).
// ============================================================================
#if USE_TDM
static __device__ inline void tdm_load_2d(void* ldsPtr, const void* gptr,
                                          unsigned tdim0, unsigned tdim1,
                                          unsigned tile0, unsigned tile1,
                                          unsigned stride0) {
    u32x4 g0;
    g0[0] = 1u;                                   // count=1, user descriptor
    g0[1] = (unsigned)(uintptr_t)ldsPtr;          // lds_addr (bytes)
    unsigned long long ga = (unsigned long long)(uintptr_t)gptr;
    g0[2] = (unsigned)ga;                         // global_addr[31:0]
    g0[3] = (unsigned)((ga >> 32) & 0x1FFFFFFu)   // global_addr[56:32]
          | (2u << 30);                           // type = 2 ("image")
    i32x8 g1;
    g1[0] = (int)(1u << 16);                      // data_size = 1 -> 2 bytes
    g1[1] = (int)((tdim0 & 0xFFFFu) << 16);       // tensor_dim0[15:0]
    g1[2] = (int)(((tdim0 >> 16) & 0xFFFFu) | ((tdim1 & 0xFFFFu) << 16));
    g1[3] = (int)(((tdim1 >> 16) & 0xFFFFu) | ((tile0 & 0xFFFFu) << 16));
    g1[4] = (int)(tile1 & 0xFFFFu);               // tile_dim1; tile_dim2=0
    g1[5] = (int)stride0;                         // tensor_dim0_stride[31:0]
    g1[6] = 0;                                    // stride0[47:32], stride1 lo
    g1[7] = 0;
    i32x4 z4 = {0, 0, 0, 0};
    i32x8 z8 = {0, 0, 0, 0, 0, 0, 0, 0};
    __builtin_amdgcn_tensor_load_to_lds(g0, g1, z4, z4, z8, 0);
}
#endif

// ============================================================================
// Kernel 2: flash attention (no 1/sqrt(d) scaling, per reference).
//   q:[B,S,H] bf16, k:[B,S,H] bf16, vT:[B,H,S] bf16 -> out:[B,S,H] f32
// Grid: (S/64, H/64, B). Block = 128 threads (4 waves).
// LDS (dynamic, 268KB): q tile 64x1024; double-buffered k tile 32x1024 and
// v tile 64x32; per-wave 16x32 probs bounce buffer. All tiles arrive via
// TDM tensor_load_to_lds (wave 0 issues, s_wait_tensorcnt 0 + barrier).
// ============================================================================
#define QS_OFF   0
#define KB_OFF   (64 * H_DIM * 2)                       // 131072
#define KB_SZ    (32 * H_DIM * 2)                       // 65536
#define VB_OFF   (KB_OFF + 2 * KB_SZ)                   // 262144
#define VB_SZ    (64 * 32 * 2)                          // 4096
#define PR_OFF   (VB_OFF + 2 * VB_SZ)                   // 270336
#define ATTN_LDS_BYTES (PR_OFF + 4 * 16 * 32 * 2)       // 274432

__global__ void attn_kernel(const __bf16* __restrict__ q,
                            const __bf16* __restrict__ k,
                            const __bf16* __restrict__ vT,
                            float* __restrict__ out) {
    extern __shared__ char smem[];
    __bf16* qs    = (__bf16*)(smem + QS_OFF);
    __bf16* probs = (__bf16*)(smem + PR_OFF);

    const int tid  = threadIdx.x;
    const int wave = tid >> 5;
    const int lane = tid & 31;
    const int l15  = lane & 15;
    const int half = (lane < 16) ? 0 : 1;

    const int q0 = blockIdx.x * 64;
    const int h0 = blockIdx.y * 64;
    const int bb = blockIdx.z;

    const __bf16* qg = q  + ((size_t)bb * S_DIM + q0) * H_DIM;
    const __bf16* kg = k  + ((size_t)bb * S_DIM) * H_DIM;        // + key*H
    const __bf16* vg = vT + ((size_t)bb * H_DIM + h0) * S_DIM;   // + key idx

    // ---------------- prologue staging: q tile + tile 0 of k/v -------------
#if USE_TDM
    if (tid < 32) {
        tdm_load_2d(smem + QS_OFF, qg, H_DIM, 64, H_DIM, 64, H_DIM);
        tdm_load_2d(smem + KB_OFF, kg, H_DIM, 32, H_DIM, 32, H_DIM);
        tdm_load_2d(smem + VB_OFF, vg, 32, 64, 32, 64, S_DIM);
        __builtin_amdgcn_s_wait_tensorcnt(0);
    }
#else
    {
        const uint4* src = (const uint4*)qg;
        uint4* dst = (uint4*)qs;
        for (int i = tid; i < (64 * H_DIM) / 8; i += 128) dst[i] = src[i];
        const uint4* ks = (const uint4*)kg;
        uint4* kd = (uint4*)(smem + KB_OFF);
        for (int i = tid; i < (32 * H_DIM) / 8; i += 128) kd[i] = ks[i];
        uint4* vd = (uint4*)(smem + VB_OFF);
        for (int i = tid; i < (64 * 32) / 8; i += 128) {
            const int r = i >> 2, c = (i & 3);
            vd[i] = ((const uint4*)(vg + (size_t)r * S_DIM))[c];
        }
    }
#endif
    __syncthreads();

    __bf16* my_probs = probs + wave * 16 * 32;

    v8f acc[4];
#pragma unroll
    for (int t = 0; t < 4; ++t)
#pragma unroll
        for (int i = 0; i < 8; ++i) acc[t][i] = 0.0f;

    float mrun[8], lrun[8];
#pragma unroll
    for (int i = 0; i < 8; ++i) { mrun[i] = -INFINITY; lrun[i] = 0.0f; }

    const __bf16* qa_base = qs + (size_t)(wave * 16 + l15) * H_DIM + half * 16;

    for (int kt = 0; kt < S_DIM / 32; ++kt) {
        const int cur = kt & 1, nxt = cur ^ 1;
        // double-buffer pointers via byte-offset arithmetic (no pointer arrays)
        __bf16* kcur = (__bf16*)(smem + KB_OFF + cur * KB_SZ);
        __bf16* vcur = (__bf16*)(smem + VB_OFF + cur * VB_SZ);
        char*   knxt = smem + KB_OFF + nxt * KB_SZ;
        char*   vnxt = smem + VB_OFF + nxt * VB_SZ;

        // ---- stage next tile (overlapped with compute in TDM path) ----
        if (kt + 1 < S_DIM / 32) {
            const int nn = (kt + 1) * 32;
#if USE_TDM
            if (tid < 32) {
                tdm_load_2d(knxt, kg + (size_t)nn * H_DIM,
                            H_DIM, 32, H_DIM, 32, H_DIM);
                tdm_load_2d(vnxt, vg + nn, 32, 64, 32, 64, S_DIM);
            }
#else
            {
                const uint4* ks = (const uint4*)(kg + (size_t)nn * H_DIM);
                uint4* kd = (uint4*)knxt;
                for (int i = tid; i < (32 * H_DIM) / 8; i += 128) kd[i] = ks[i];
                uint4* vd = (uint4*)vnxt;
                for (int i = tid; i < (64 * 32) / 8; i += 128) {
                    const int r = i >> 2, c = (i & 3);
                    vd[i] = ((const uint4*)(vg + nn + (size_t)r * S_DIM))[c];
                }
            }
#endif
        }

        // ---------- scores: 16 q-rows x 32 keys, K = 1024 (from LDS) -------
        v8f s0, s1;
#pragma unroll
        for (int i = 0; i < 8; ++i) { s0[i] = 0.0f; s1[i] = 0.0f; }

        const __bf16* kb0 = kcur + (size_t)l15 * H_DIM + half * 16;
        const __bf16* kb1 = kb0 + (size_t)16 * H_DIM;

        for (int kk = 0; kk < H_DIM; kk += 32) {
            v16bf a  = load_a_frag_bf16(qa_base + kk);
            v16bf b0 = load_b_frag_bf16(kb0 + kk);
            v16bf b1 = load_b_frag_bf16(kb1 + kk);
            s0 = wmma_acc(a, b0, s0);
            s1 = wmma_acc(a, b1, s1);
        }

        // ---------- online softmax ----------
#pragma unroll
        for (int i = 0; i < 8; ++i) {
            float t = fmaxf(s0[i], s1[i]);
            t = fmaxf(t, __shfl_xor(t, 1, 32));
            t = fmaxf(t, __shfl_xor(t, 2, 32));
            t = fmaxf(t, __shfl_xor(t, 4, 32));
            t = fmaxf(t, __shfl_xor(t, 8, 32));
            const float mnew = fmaxf(mrun[i], t);
            const float f    = __expf(mrun[i] - mnew);   // 0 on first tile
            mrun[i] = mnew;

            const float p0 = __expf(s0[i] - mnew);
            const float p1 = __expf(s1[i] - mnew);
            float r = p0 + p1;
            r += __shfl_xor(r, 1, 32);
            r += __shfl_xor(r, 2, 32);
            r += __shfl_xor(r, 4, 32);
            r += __shfl_xor(r, 8, 32);
            lrun[i] = lrun[i] * f + r;

#pragma unroll
            for (int t4 = 0; t4 < 4; ++t4) acc[t4][i] *= f;

            const int prow = i + half * 8;               // C-layout -> LDS
            my_probs[prow * 32 + l15]      = (__bf16)p0;
            my_probs[prow * 32 + 16 + l15] = (__bf16)p1;
        }
        __syncthreads();   // probs writes visible as A-fragments

        // ---------- PV: context += probs(16x32) @ v(32 x 64-slice) ---------
        v16bf afrag = load_a_frag_bf16(my_probs + l15 * 32 + half * 16);
#pragma unroll
        for (int t = 0; t < 4; ++t) {
            const __bf16* vb = vcur + (size_t)(t * 16 + l15) * 32 + half * 16;
            v16bf bfrag = load_b_frag_bf16(vb);
            acc[t] = wmma_acc(afrag, bfrag, acc[t]);
        }

#if USE_TDM
        if (tid < 32) __builtin_amdgcn_s_wait_tensorcnt(0);
#endif
        __syncthreads();   // next-tile data ready; probs WAR; buffer swap safe
    }

    // ---------- normalize + store ----------
#pragma unroll
    for (int t = 0; t < 4; ++t) {
        const int h = h0 + t * 16 + l15;
#pragma unroll
        for (int i = 0; i < 8; ++i) {
            const int row = q0 + wave * 16 + i + half * 8;
            out[((size_t)bb * S_DIM + row) * H_DIM + h] = acc[t][i] / lrun[i];
        }
    }
}

// ============================================================================
extern "C" void kernel_launch(void* const* d_in, const int* in_sizes, int n_in,
                              void* d_out, int out_size, void* d_ws, size_t ws_size,
                              hipStream_t stream) {
    const float* x  = (const float*)d_in[0];
    const float* Wq = (const float*)d_in[1];
    const float* bq = (const float*)d_in[2];
    const float* Wk = (const float*)d_in[3];
    const float* bk = (const float*)d_in[4];
    const float* Wv = (const float*)d_in[5];
    const float* bv = (const float*)d_in[6];
    float* out = (float*)d_out;

    const size_t MH = (size_t)M_DIM * H_DIM;   // 16.7M elems
    const size_t HH = (size_t)H_DIM * H_DIM;   // 1.05M elems
    __bf16* qws  = (__bf16*)d_ws;
    __bf16* kws  = qws  + MH;
    __bf16* vtws = kws  + MH;
    __bf16* xb   = vtws + MH;
    __bf16* wqb  = xb   + MH;
    __bf16* wkb  = wqb  + HH;
    __bf16* wvb  = wkb  + HH;

    // ---- one-shot fp32 -> bf16 conversions ----
    {
        const int n8x = (int)(MH / 8), n8w = (int)(HH / 8);
        cvt_bf16_kernel<<<(n8x + 255) / 256, 256, 0, stream>>>(x,  xb,  n8x);
        cvt_bf16_kernel<<<(n8w + 255) / 256, 256, 0, stream>>>(Wq, wqb, n8w);
        cvt_bf16_kernel<<<(n8w + 255) / 256, 256, 0, stream>>>(Wk, wkb, n8w);
        cvt_bf16_kernel<<<(n8w + 255) / 256, 256, 0, stream>>>(Wv, wvb, n8w);
    }

    dim3 gemm_grid(M_DIM / 64, H_DIM / 64);
    dim3 blk(128);
    qkv_proj_kernel<<<gemm_grid, blk, 0, stream>>>(xb, wqb, bq, qws, 0);
    qkv_proj_kernel<<<gemm_grid, blk, 0, stream>>>(xb, wkb, bk, kws, 0);
    qkv_proj_kernel<<<gemm_grid, blk, 0, stream>>>(xb, wvb, bv, vtws, 1);

    dim3 attn_grid(S_DIM / 64, H_DIM / 64, B_DIM);
    attn_kernel<<<attn_grid, blk, ATTN_LDS_BYTES, stream>>>(qws, kws, vtws, out);
}
